// Shapenet_34076270526641
// MI455X (gfx1250) — matrix operand order repaired and checked by reading, not compiled, not run on hardware
//
#include <hip/hip_runtime.h>
#include <hip/hip_bf16.h>
#include <math.h>

// ---------------------------------------------------------------------------
// MI455X (gfx1250) implementation notes:
//  * Dominant cost: E=200k per-edge 32x32 matvecs with edge-conditioned
//    weights EW[e] = hid[e,:8] @ e2_w (8x1024). Materializing EW = 819MB ->
//    ~3.3GB HBM traffic over 3 rounds (~140us @ 23.3TB/s). Instead use
//      m[e,o] = sum_k hid[e,k] * (hsrc[e,:] @ W2r[k])[o] + (hsrc[e,:] @ B2r)[o]
//    so the 32-wide hsrc contraction runs on v_wmma_f32_16x16x32_bf16 at the
//    NATIVE K=32 shape (no padding, both lane halves dense) and only the k=8
//    contraction is VALU. 18 dense WMMAs + ~144 lane-FMAs per 16-edge tile;
//    EW never exists in memory.
//  * All WMMA operands are pre-packed in LDS in exact operand register layout
//    (bf16 pairs in dwords, [chunk][lane][vgpr]) so each operand build is two
//    ds_load_b128 instead of 8+ ds_load_u16.
//  * Node update GEMMs (root_w, GRU) are K=32 == native WMMA shape; one wave
//    per 16-node tile, weights pre-packed bf16 in LDS.
//  * Scatter-add uses global_atomic_add_f32 into agg (2.5MB, L2-resident on
//    a 192MB L2).
//  * Wave-local LDS producer->consumer uses s_wait_dscnt 0 (CDNA5 split
//    counters) instead of workgroup barriers.
// ---------------------------------------------------------------------------

#define HH 32
#define BB 16

typedef __attribute__((ext_vector_type(16))) __bf16 v16bf;
typedef __attribute__((ext_vector_type(8)))  float  v8f;

__device__ __forceinline__ v8f wmma_bf16(v16bf a, v16bf b, v8f c) {
  // (neg_a, A, neg_b, B, c_mod, C, reuse_a, reuse_b)
  return __builtin_amdgcn_wmma_f32_16x16x32_bf16(false, a, false, b,
                                                 (short)0, c, false, false);
}

__device__ __forceinline__ float sigm(float x) { return 1.0f / (1.0f + expf(-x)); }

__device__ __forceinline__ unsigned int packbf(float a, float b) {
  union { __bf16 h[2]; unsigned int u; } t;
  t.h[0] = (__bf16)a;
  t.h[1] = (__bf16)b;
  return t.u;
}

// Load a pre-packed 8-dword (32B-aligned) WMMA operand from LDS.
__device__ __forceinline__ v16bf ldOp(const unsigned int* p) {
  return *reinterpret_cast<const v16bf*>(p);
}

// A-operand K index for packed dword r (contiguous order, matches B packing):
// VGPR r: K=2r,2r+1 (+16 upper half).
__device__ __forceinline__ int kbaseA(int r) {
  return 2 * r;
}
// B-operand K index for packed dword r (contiguous order, matches A packing):
// VGPR r: K=2r,2r+1 lanes 0-15; K=16+2r,16+2r+1 lanes 16-31.

// A-operand built from an f32 row tile in LDS (node kernel path).
__device__ __forceinline__ v16bf makeA(const float* rowTile /*16x32 f32*/, int lane) {
  int m = lane & 15, hf = lane >> 4;
  v16bf A;
#pragma unroll
  for (int r = 0; r < 8; ++r) {
    int k = kbaseA(r) + 16 * hf;
    A[2 * r]     = (__bf16)rowTile[m * 32 + k];
    A[2 * r + 1] = (__bf16)rowTile[m * 32 + k + 1];
  }
  return A;
}

// ---------------------------------------------------------------------------
__global__ __launch_bounds__(256) void proj_kernel(
    const float* __restrict__ x, const float* __restrict__ w,
    const float* __restrict__ b, float* __restrict__ h, int N_) {
  int i = blockIdx.x * 256 + threadIdx.x;
  if (i >= N_ * HH) return;
  int n = i >> 5, c = i & 31;
  float acc = b[c];
#pragma unroll
  for (int k = 0; k < 4; ++k) acc += x[n * 4 + k] * w[k * HH + c];
  h[i] = fmaxf(acc, 0.0f);
}

__global__ __launch_bounds__(256) void ehid_kernel(
    const float* __restrict__ ea, const float* __restrict__ w,
    const float* __restrict__ b, float* __restrict__ hid, int E_) {
  int i = blockIdx.x * 256 + threadIdx.x;
  if (i >= E_ * 8) return;
  int e = i >> 3, k = i & 7;
  hid[i] = fmaxf(ea[e] * w[k] + b[k], 0.0f);
}

__global__ __launch_bounds__(256) void zero_kernel(float* __restrict__ p, int n) {
  int i = blockIdx.x * 256 + threadIdx.x;
  if (i < n) p[i] = 0.0f;
}

// ---------------------------------------------------------------------------
// Fused edge-weight-generation + per-edge matvec + scatter-add.
// One wave per 16-edge tile; 8 tiles per 256-thread block.
// WMMA contracts the 32-wide hsrc dim (native shape); k=8 contraction in VALU.
__global__ __launch_bounds__(256) void msg_kernel(
    const float* __restrict__ h,    // N x 32
    const float* __restrict__ hid,  // E x 8
    const float* __restrict__ e2w,  // 8 x 1024  (= [k][i*32+o])
    const float* __restrict__ e2b,  // 1024      (= [i*32+o])
    const int* __restrict__ src, const int* __restrict__ dst,
    float* __restrict__ agg,        // N x 32 (atomic)
    int E_) {
  // B operands pre-packed in exact WMMA layout: [slot][lane][vgpr dword].
  __shared__ unsigned int sW2p[8 * 2 * 32 * 8];  // 16 KB: W2r[k], o-chunk c
  __shared__ unsigned int sB2p[2 * 32 * 8];      //  2 KB: bias matrix B2r
  __shared__ unsigned int sAT[8][2 * 16 * 8];    //  8 KB: per-wave packed A
  __shared__ float        sHid[8][8 * 16];       //  4 KB: per-wave hid^T [k][e]
  int tid = threadIdx.x;

  // Pack W2r[k][i][o] = e2w[k*1024 + i*32 + o] into B layout.
  for (int idx = tid; idx < 8 * 2 * 32 * 8; idx += 256) {
    int r = idx & 7, lane_ = (idx >> 3) & 31, c = (idx >> 8) & 1, k = idx >> 9;
    int nn_ = lane_ & 15, hf_ = lane_ >> 4;
    int K = 2 * r + 16 * hf_, o = c * 16 + nn_;
    sW2p[idx] = packbf(e2w[k * 1024 + K * 32 + o],
                       e2w[k * 1024 + (K + 1) * 32 + o]);
  }
  for (int idx = tid; idx < 2 * 32 * 8; idx += 256) {
    int r = idx & 7, lane_ = (idx >> 3) & 31, c = idx >> 8;
    int nn_ = lane_ & 15, hf_ = lane_ >> 4;
    int K = 2 * r + 16 * hf_, o = c * 16 + nn_;
    sB2p[idx] = packbf(e2b[K * 32 + o], e2b[(K + 1) * 32 + o]);
  }
  __syncthreads();

  int wave = tid >> 5, lane = tid & 31;
  int half = lane >> 4, nn = lane & 15;
  int tile = blockIdx.x * 8 + wave;
  if (tile * 16 >= E_) return;
  int e0 = tile * 16;

  // Stage packed A (hsrc tile, 16x32 -> bf16 pairs in operand order) and
  // hid^T tile for this wave.
  unsigned int* aT = &sAT[wave][0];
  float* hT = &sHid[wave][0];
  for (int idx = lane; idx < 256; idx += 32) {
    int r = idx & 7, m = (idx >> 3) & 15, hf_ = idx >> 7;
    int ge = e0 + m; ge = ge < E_ ? ge : E_ - 1;
    const float* hr = &h[(size_t)src[ge] * HH];
    int K = kbaseA(r) + 16 * hf_;
    aT[idx] = packbf(hr[K], hr[K + 1]);
  }
  for (int idx = lane; idx < 128; idx += 32) {
    int k = idx >> 4, er = idx & 15;
    int ge = e0 + er; ge = ge < E_ ? ge : E_ - 1;
    hT[idx] = hid[(size_t)ge * 8 + k];
  }
  asm volatile("s_wait_dscnt 0" ::: "memory");  // wave-local LDS handoff

  v16bf A = ldOp(&aT[(half * 16 + nn) * 8]);  // reused by all 18 WMMAs

  float mac[2][8];
#pragma unroll
  for (int c = 0; c < 2; ++c) {  // bias term: hsrc @ B2r
    v8f Cb = {};
    Cb = wmma_bf16(A, ldOp(&sB2p[(c * 32 + lane) * 8]), Cb);
#pragma unroll
    for (int r = 0; r < 8; ++r) mac[c][r] = Cb[r];
  }
#pragma unroll
  for (int k = 0; k < 8; ++k) {
    // hid broadcast values for this k (rows of this lane half).
    float4 hv0 = *(const float4*)&hT[k * 16 + 8 * half];
    float4 hv1 = *(const float4*)&hT[k * 16 + 8 * half + 4];
    float hv[8] = {hv0.x, hv0.y, hv0.z, hv0.w, hv1.x, hv1.y, hv1.z, hv1.w};
#pragma unroll
    for (int c = 0; c < 2; ++c) {
      v8f C = {};
      C = wmma_bf16(A, ldOp(&sW2p[((k * 2 + c) * 32 + lane) * 8]), C);
#pragma unroll
      for (int r = 0; r < 8; ++r) mac[c][r] += hv[r] * C[r];
    }
  }

#pragma unroll
  for (int r = 0; r < 8; ++r) {
    int ge = e0 + r + 8 * half;  // C row M = r + 8*half
    if (ge < E_) {
      int d = dst[ge];
      atomicAdd(&agg[(size_t)d * HH + nn],      mac[0][r]);
      atomicAdd(&agg[(size_t)d * HH + 16 + nn], mac[1][r]);
    }
  }
}

// ---------------------------------------------------------------------------
// Node update: relu(agg + h@root_w + b) -> GRU.  One wave per 16-node tile.
// All GEMMs K=32 == native v_wmma_f32_16x16x32_bf16 shape.
__global__ __launch_bounds__(256) void node_kernel(
    const float* __restrict__ h, const float* __restrict__ agg,
    const float* __restrict__ root_w, const float* __restrict__ conv_b,
    const float* __restrict__ gru_wi, const float* __restrict__ gru_wh,
    const float* __restrict__ gru_bi, const float* __restrict__ gru_bh,
    float* __restrict__ hNext, int N_) {
  __shared__ unsigned int sRootP[2 * 32 * 8];  // 2 KB packed B operands
  __shared__ unsigned int sWiP[6 * 32 * 8];    // 6 KB (B[k][c] = wi[c][k])
  __shared__ unsigned int sWhP[6 * 32 * 8];    // 6 KB
  __shared__ float sCb[32], sBi[96], sBh[96];
  __shared__ float sHT[8][16 * 32];  // 16 KB
  __shared__ float sOT[8][16 * 32];  // 16 KB
  int tid = threadIdx.x;
  for (int idx = tid; idx < 2 * 32 * 8; idx += 256) {
    int r = idx & 7, lane_ = (idx >> 3) & 31, c = idx >> 8;
    int nn_ = lane_ & 15, hf_ = lane_ >> 4;
    int K = 2 * r + 16 * hf_, o = c * 16 + nn_;
    sRootP[idx] = packbf(root_w[K * 32 + o], root_w[(K + 1) * 32 + o]);
  }
  for (int idx = tid; idx < 6 * 32 * 8; idx += 256) {
    int r = idx & 7, lane_ = (idx >> 3) & 31, c = idx >> 8;
    int nn_ = lane_ & 15, hf_ = lane_ >> 4;
    int K = 2 * r + 16 * hf_, o = c * 16 + nn_;
    // B[k][c] = gru_w*[c][k] -> pair (K,K+1) is contiguous in gru_w* row o.
    sWiP[idx] = packbf(gru_wi[o * 32 + K], gru_wi[o * 32 + K + 1]);
    sWhP[idx] = packbf(gru_wh[o * 32 + K], gru_wh[o * 32 + K + 1]);
  }
  if (tid < 32) sCb[tid] = conv_b[tid];
  if (tid < 96) { sBi[tid] = gru_bi[tid]; sBh[tid] = gru_bh[tid]; }
  __syncthreads();

  int wave = tid >> 5, lane = tid & 31;
  int half = lane >> 4, nn = lane & 15;
  int tile = blockIdx.x * 8 + wave;
  if (tile * 16 >= N_) return;
  int n0 = tile * 16;
  float* hT = &sHT[wave][0];
  float* oT = &sOT[wave][0];
  for (int i = lane; i < 16 * 32; i += 32) {
    int nd = n0 + (i >> 5);
    hT[i] = (nd < N_) ? h[(size_t)nd * HH + (i & 31)] : 0.0f;
  }
  asm volatile("s_wait_dscnt 0" ::: "memory");

  v16bf Ah = makeA(hT, lane);

  // conv: out = relu(agg + h @ root_w + b)
#pragma unroll
  for (int ci = 0; ci < 2; ++ci) {
    int c0 = ci * 16;
    v8f C = {};
    C = wmma_bf16(Ah, ldOp(&sRootP[(ci * 32 + lane) * 8]), C);
#pragma unroll
    for (int r = 0; r < 8; ++r) {
      int m = r + 8 * half, nd = n0 + m, c = c0 + nn;
      float av = (nd < N_) ? agg[(size_t)nd * HH + c] : 0.0f;
      oT[m * 32 + c] = fmaxf(av + C[r] + sCb[c], 0.0f);
    }
  }
  asm volatile("s_wait_dscnt 0" ::: "memory");
  v16bf Ao = makeA(oT, lane);

  // GRU: gi = out@wi^T + bi ; gh = msg@wh^T + bh (msg == h this round)
  float rg[2][8], zg[2][8], ng[2][8];
#pragma unroll
  for (int cc = 0; cc < 6; ++cc) {
    int c0 = cc * 16;
    v8f Ci = {}; Ci = wmma_bf16(Ao, ldOp(&sWiP[(cc * 32 + lane) * 8]), Ci);
    v8f Ch = {}; Ch = wmma_bf16(Ah, ldOp(&sWhP[(cc * 32 + lane) * 8]), Ch);
    float bi = sBi[c0 + nn], bh = sBh[c0 + nn];
#pragma unroll
    for (int r = 0; r < 8; ++r) {
      float gi = Ci[r] + bi, gh = Ch[r] + bh;
      if (cc < 2)      rg[cc][r]     = sigm(gi + gh);
      else if (cc < 4) zg[cc - 2][r] = sigm(gi + gh);
      else             ng[cc - 4][r] = tanhf(gi + rg[cc - 4][r] * gh);
    }
  }
#pragma unroll
  for (int ci = 0; ci < 2; ++ci)
#pragma unroll
    for (int r = 0; r < 8; ++r) {
      int m = r + 8 * half, c = ci * 16 + nn, nd = n0 + m;
      if (nd < N_) {
        float msgv = hT[m * 32 + c];
        hNext[(size_t)nd * HH + c] =
            (1.0f - zg[ci][r]) * ng[ci][r] + zg[ci][r] * msgv;
      }
    }
}

// ---------------------------------------------------------------------------
// Set2Set with zero initial state collapses exactly: g = lstm_bi + lstm_bh.
__global__ void q_kernel(const float* __restrict__ bi, const float* __restrict__ bh,
                         float* __restrict__ q) {
  int tid = threadIdx.x;
  if (tid >= BB * HH) return;
  int c = tid & 31;
  float gi = bi[c] + bh[c];
  float gg = bi[64 + c] + bh[64 + c];
  float go = bi[96 + c] + bh[96 + c];
  float cl = sigm(gi) * tanhf(gg);  // f gate multiplies cL==0
  q[tid] = sigm(go) * tanhf(cl);
}

__global__ __launch_bounds__(256) void att_e_kernel(
    const float* __restrict__ h, const float* __restrict__ q,
    const int* __restrict__ batch, float* __restrict__ e, int N_) {
  int nd = blockIdx.x * 256 + threadIdx.x;
  if (nd >= N_) return;
  const float* qb = &q[(size_t)batch[nd] * HH];
  float acc = 0.0f;
#pragma unroll
  for (int c = 0; c < HH; ++c) acc += h[(size_t)nd * HH + c] * qb[c];
  e[nd] = acc;
}

__global__ __launch_bounds__(256) void seg_max_kernel(
    const float* __restrict__ e, const int* __restrict__ batch,
    float* __restrict__ emax, float* __restrict__ asum,
    float* __restrict__ rout, int N_) {
  __shared__ float red[256];
  int b = blockIdx.x, tid = threadIdx.x;
  float mx = -3.0e38f;
  for (int nd = tid; nd < N_; nd += 256)
    if (batch[nd] == b) mx = fmaxf(mx, e[nd]);
  red[tid] = mx;
  __syncthreads();
  for (int s = 128; s > 0; s >>= 1) {
    if (tid < s) red[tid] = fmaxf(red[tid], red[tid + s]);
    __syncthreads();
  }
  if (tid == 0) { emax[b] = red[0]; asum[b] = 0.0f; }
  if (tid < HH) rout[b * HH + tid] = 0.0f;
}

__global__ __launch_bounds__(256) void att_a_kernel(
    float* __restrict__ e, const int* __restrict__ batch,
    const float* __restrict__ emax, float* __restrict__ asum, int N_) {
  int nd = blockIdx.x * 256 + threadIdx.x;
  if (nd >= N_) return;
  int b = batch[nd];
  float a = expf(e[nd] - emax[b]);
  e[nd] = a;  // reuse buffer as 'a'
  atomicAdd(&asum[b], a);
}

__global__ __launch_bounds__(256) void rout_kernel(
    const float* __restrict__ a, const int* __restrict__ batch,
    const float* __restrict__ asum, const float* __restrict__ h,
    float* __restrict__ rout, int N_) {
  int i = blockIdx.x * 256 + threadIdx.x;
  if (i >= N_ * HH) return;
  int nd = i >> 5, c = i & 31;
  int b = batch[nd];
  atomicAdd(&rout[b * HH + c], (a[nd] / asum[b]) * h[(size_t)nd * HH + c]);
}

__global__ void mlp_kernel(const float* __restrict__ q, const float* __restrict__ rout,
                           const float* __restrict__ f1w, const float* __restrict__ f1b,
                           const float* __restrict__ f2w, const float* __restrict__ f2b,
                           const float* __restrict__ f3w, const float* __restrict__ f3b,
                           float* __restrict__ out) {
  __shared__ float qs[BB][2 * HH];
  __shared__ float o1[BB][HH];
  __shared__ float o2[BB][HH];
  int tid = threadIdx.x;
  int b = tid >> 5, c = tid & 31;
  qs[b][c] = q[b * HH + c];
  qs[b][HH + c] = rout[b * HH + c];
  __syncthreads();
  float acc = f1b[c];
  for (int k = 0; k < 2 * HH; ++k) acc += qs[b][k] * f1w[k * HH + c];
  o1[b][c] = fmaxf(acc, 0.0f);
  __syncthreads();
  acc = f2b[c];
  for (int k = 0; k < HH; ++k) acc += o1[b][k] * f2w[k * HH + c];
  o2[b][c] = fmaxf(acc, 0.0f);
  __syncthreads();
  if (c == 0) {
    float s = f3b[0];
    for (int k = 0; k < HH; ++k) s += o2[b][k] * f3w[k];
    out[b] = s;
  }
}

// ---------------------------------------------------------------------------
extern "C" void kernel_launch(void* const* d_in, const int* in_sizes, int n_in,
                              void* d_out, int out_size, void* d_ws, size_t ws_size,
                              hipStream_t stream) {
  const float* x      = (const float*)d_in[0];
  const int*   ei     = (const int*)d_in[1];
  const float* ea     = (const float*)d_in[2];
  const int*   batch  = (const int*)d_in[3];
  const float* proj_w = (const float*)d_in[4];
  const float* proj_b = (const float*)d_in[5];
  const float* e1_w   = (const float*)d_in[6];
  const float* e1_b   = (const float*)d_in[7];
  const float* e2_w   = (const float*)d_in[8];
  const float* e2_b   = (const float*)d_in[9];
  const float* root_w = (const float*)d_in[10];
  const float* conv_b = (const float*)d_in[11];
  const float* gru_wi = (const float*)d_in[12];
  const float* gru_wh = (const float*)d_in[13];
  const float* gru_bi = (const float*)d_in[14];
  const float* gru_bh = (const float*)d_in[15];
  const float* lstm_bi = (const float*)d_in[18];
  const float* lstm_bh = (const float*)d_in[19];
  const float* f1_w = (const float*)d_in[20];
  const float* f1_b = (const float*)d_in[21];
  const float* f2_w = (const float*)d_in[22];
  const float* f2_b = (const float*)d_in[23];
  const float* f3_w = (const float*)d_in[24];
  const float* f3_b = (const float*)d_in[25];

  int N = in_sizes[0] / 4;
  int E = in_sizes[2];
  const int* src = ei;
  const int* dst = ei + E;

  // workspace layout (f32): ~14.3 MB total
  float* ws    = (float*)d_ws;
  float* h0    = ws;
  float* h1    = h0 + (size_t)N * HH;
  float* hid   = h1 + (size_t)N * HH;
  float* agg   = hid + (size_t)E * 8;
  float* ebuf  = agg + (size_t)N * HH;
  float* qbuf  = ebuf + N;
  float* emaxb = qbuf + BB * HH;
  float* asum  = emaxb + BB;
  float* rout  = asum + BB;

  proj_kernel<<<(N * HH + 255) / 256, 256, 0, stream>>>(x, proj_w, proj_b, h0, N);
  ehid_kernel<<<(E * 8 + 255) / 256, 256, 0, stream>>>(ea, e1_w, e1_b, hid, E);

  int tilesE = (E + 15) / 16, blkE = (tilesE + 7) / 8;
  int tilesN = (N + 15) / 16, blkN = (tilesN + 7) / 8;
  float* hc = h0;
  float* hn = h1;
  for (int r = 0; r < 3; ++r) {
    zero_kernel<<<(N * HH + 255) / 256, 256, 0, stream>>>(agg, N * HH);
    msg_kernel<<<blkE, 256, 0, stream>>>(hc, hid, e2_w, e2_b, src, dst, agg, E);
    node_kernel<<<blkN, 256, 0, stream>>>(hc, agg, root_w, conv_b, gru_wi, gru_wh,
                                          gru_bi, gru_bh, hn, N);
    float* t = hc; hc = hn; hn = t;
  }

  q_kernel<<<1, 512, 0, stream>>>(lstm_bi, lstm_bh, qbuf);
  att_e_kernel<<<(N + 255) / 256, 256, 0, stream>>>(hc, qbuf, batch, ebuf, N);
  seg_max_kernel<<<BB, 256, 0, stream>>>(ebuf, batch, emaxb, asum, rout, N);
  att_a_kernel<<<(N + 255) / 256, 256, 0, stream>>>(ebuf, batch, emaxb, asum, N);
  rout_kernel<<<(N * HH + 255) / 256, 256, 0, stream>>>(ebuf, batch, asum, hc, rout, N);
  mlp_kernel<<<1, 512, 0, stream>>>(qbuf, rout, f1_w, f1_b, f2_w, f2_b, f3_w, f3_b,
                                    (float*)d_out);
}
